// LSTMModel_55611236548767
// MI455X (gfx1250) — matrix operand order repaired and checked by reading, not compiled, not run on hardware
//
#include <hip/hip_runtime.h>

// LSTM: B=1024, T=336, I=16, H=64, 2 layers + FC(64->1)
#define TLEN 336
#define IDIM 16
#define HDIM 64

typedef float  v8f   __attribute__((ext_vector_type(8)));
typedef __bf16 v16bf __attribute__((ext_vector_type(16)));

__global__ __launch_bounds__(128, 1)
void lstm_fused_kernel(const float* __restrict__ x,
                       const float* __restrict__ Wih0, const float* __restrict__ Whh0,
                       const float* __restrict__ bih0, const float* __restrict__ bhh0,
                       const float* __restrict__ Wih1, const float* __restrict__ Whh1,
                       const float* __restrict__ bih1, const float* __restrict__ bhh1,
                       const float* __restrict__ Wfc,  const float* __restrict__ bfc,
                       float* __restrict__ out)
{
#if defined(__HIP_DEVICE_COMPILE__)
  constexpr int APAD = 24;                 // u16 stride per A-frag lane (16 data + 8 pad -> 48B, bank-spread)
  constexpr int KBLK = 32 * APAD;          // one 16x32 A fragment (32 lanes)
  constexpr float L2E  = 1.4426950408889634f;

  __shared__ __align__(16) unsigned short xA[32 * APAD];        // x[t] A-frag (K=16 live, K>=16 zero)
  __shared__ __align__(16) unsigned short h0A[2][2 * KBLK];     // layer0 h, double buffered, 2 k-blocks
  __shared__ __align__(16) unsigned short h1A[2][2 * KBLK];     // layer1 h
  __shared__ float hOut[16 * 64];                               // final h1 in f32 for FC

  const int tid  = threadIdx.x;
  const int lane = tid & 31;
  const int j    = tid >> 5;               // wave id == hidden column group (h in [16j,16j+16))
  const int b0   = blockIdx.x * 16;        // batch tile base

  // ---- zero LDS staging (padding + K-pad region of xA stays zero forever) ----
  for (int i = tid; i < 32 * APAD; i += 128) xA[i] = 0;
  for (int i = tid; i < 2 * KBLK; i += 128) {
    h0A[0][i] = 0; h0A[1][i] = 0; h1A[0][i] = 0; h1A[1][i] = 0;
  }

  // ---- helpers ----
  union Frag { v16bf v; unsigned short u[16]; uint4 q[2]; };
  auto f2bf = [](float f) -> unsigned short {
    union { float f; unsigned int u; } c; c.f = f;
    unsigned int r = c.u + 0x7fffu + ((c.u >> 16) & 1u);     // round-to-nearest-even
    return (unsigned short)(r >> 16);
  };

  // ---- build all weight B-fragments in registers (used 336x each) ----
  // B layout (16-bit 32x16): lane l holds column N=(l&15); K = 16*(e>>3) + 8*(l>>4) + (e&7), e=0..15.
  const int nl   = lane & 15;
  const int half = lane >> 4;

  v16bf wih0[4], whh0[2][4], wih1[2][4], whh1[2][4];
  // biases pre-scaled into the activation argument:
  //   sigmoid gates (i,f,o): exp2 arg = -L2E*v + sb,  sb = -L2E*bias
  //   tanh gate (g):         exp2 arg =  2*L2E*v + tb, tb = 2*L2E*bias
  float sb0[4], sb1[4];
  #pragma unroll
  for (int gi = 0; gi < 4; ++gi) {                 // gate: 0=i 1=f 2=g 3=o
    const int n = (gi * 4 + j) * 16 + nl;          // output column in [0,256)
    const float b0s = bih0[n] + bhh0[n];
    const float b1s = bih1[n] + bhh1[n];
    const float scale = (gi == 2) ? (2.0f * L2E) : (-L2E);
    sb0[gi] = scale * b0s;
    sb1[gi] = scale * b1s;
    { // W_ih0 [256][16] -> K padded to 32 with zeros
      Frag f;
      #pragma unroll
      for (int e = 0; e < 16; ++e) {
        int k = half * 8 + e + 8 * (e >> 3);
        f.u[e] = (k < IDIM) ? f2bf(Wih0[n * IDIM + k]) : (unsigned short)0;
      }
      wih0[gi] = f.v;
    }
    #pragma unroll
    for (int kb = 0; kb < 2; ++kb) {               // K = 64 -> two 32-blocks
      Frag a, b, c;
      #pragma unroll
      for (int e = 0; e < 16; ++e) {
        int k = kb * 32 + half * 8 + e + 8 * (e >> 3);
        a.u[e] = f2bf(Whh0[n * HDIM + k]);
        b.u[e] = f2bf(Wih1[n * HDIM + k]);
        c.u[e] = f2bf(Whh1[n * HDIM + k]);
      }
      whh0[kb][gi] = a.v; wih1[kb][gi] = b.v; whh1[kb][gi] = c.v;
    }
  }

  // ---- x staging: each thread feeds 2 consecutive floats (bl, i0..i0+1) ----
  // A layout (16-bit 16x32): element (l',e) holds M=l'&15, K = 16*(e>>3) + 8*(l'>>4) + (e&7).
  const int bl = tid >> 3;                         // local batch row 0..15
  const int i0 = (tid & 7) * 2;                    // input feature 0,2,...,14
  const float* xptr = x + ((size_t)(b0 + bl) * TLEN) * IDIM + i0;
  const int xIdx = (bl + 16 * ((i0 >> 3) & 1)) * APAD + (i0 & 7);

  // ---- elementwise->A-frag scatter base (8 values / lane, stride APAD) ----
  const int stBase = (j >> 1) * KBLK
                   + (8 * (lane >> 4) + 16 * ((lane >> 3) & 1)) * APAD
                   + 8 * (j & 1) + (lane & 7);

  // sigmoid(v + b) with sb = -L2E*b
  auto sigm_b = [](float v, float sb) -> float {
    return __builtin_amdgcn_rcpf(1.0f + __builtin_amdgcn_exp2f(__builtin_fmaf(-L2E, v, sb)));
  };
  // tanh(v + b) with tb = 2*L2E*b
  auto tanh_b = [](float v, float tb) -> float {
    return 1.0f - 2.0f * __builtin_amdgcn_rcpf(1.0f + __builtin_amdgcn_exp2f(__builtin_fmaf(2.0f * L2E, v, tb)));
  };
  // plain tanh(c)
  auto tanh_f = [](float v) -> float {
    return 1.0f - 2.0f * __builtin_amdgcn_rcpf(1.0f + __builtin_amdgcn_exp2f(2.0f * L2E * v));
  };

  v8f c0v = {0.f,0.f,0.f,0.f,0.f,0.f,0.f,0.f};
  v8f c1v = {0.f,0.f,0.f,0.f,0.f,0.f,0.f,0.f};

  // preload x[0]
  {
    float2 xv = *(const float2*)xptr;
    unsigned int p = (unsigned int)f2bf(xv.x) | ((unsigned int)f2bf(xv.y) << 16);
    *(unsigned int*)&xA[xIdx] = p;
  }
  __syncthreads();

  for (int t = 0; t < TLEN; ++t) {
    const int pr = t & 1, pw = pr ^ 1;
    v8f acc[4];

    // ===== layer 0 gates: x@Wih0^T + h0@Whh0^T  (bias folded into activations) =====
    #pragma unroll
    for (int gi = 0; gi < 4; ++gi) acc[gi] = (v8f){0.f,0.f,0.f,0.f,0.f,0.f,0.f,0.f};
    {
      Frag ax;
      ax.q[0] = *(const uint4*)&xA[lane * APAD];
      ax.q[1] = *(const uint4*)&xA[lane * APAD + 8];
      #pragma unroll
      for (int gi = 0; gi < 4; ++gi)
        acc[gi] = __builtin_amdgcn_wmma_f32_16x16x32_bf16(false, ax.v, false, wih0[gi],
                                                          (short)0, acc[gi], false, false);
    }
    #pragma unroll
    for (int kb = 0; kb < 2; ++kb) {
      Frag ah;
      const unsigned short* s = &h0A[pr][kb * KBLK + lane * APAD];
      ah.q[0] = *(const uint4*)s; ah.q[1] = *(const uint4*)(s + 8);
      #pragma unroll
      for (int gi = 0; gi < 4; ++gi)
        acc[gi] = __builtin_amdgcn_wmma_f32_16x16x32_bf16(false, ah.v, false, whh0[kb][gi],
                                                          (short)0, acc[gi], false, false);
    }
    // elementwise LSTM cell, layer 0 (in WMMA C layout)
    #pragma unroll
    for (int v = 0; v < 8; ++v) {
      float ig = sigm_b(acc[0][v], sb0[0]);
      float fg = sigm_b(acc[1][v], sb0[1]);
      float gg = tanh_b(acc[2][v], sb0[2]);
      float og = sigm_b(acc[3][v], sb0[3]);
      float c  = fg * c0v[v] + ig * gg;
      c0v[v]   = c;
      float h  = og * tanh_f(c);
      h0A[pw][stBase + v * APAD] = f2bf(h);
    }
    __syncthreads();   // h0A[pw] visible to all waves

    // issue next-step x load early so VMEM latency overlaps the layer-1 WMMAs
    float2 xv = {0.f, 0.f};
    if (t + 1 < TLEN)
      xv = *(const float2*)(xptr + (size_t)(t + 1) * IDIM);
    if (t + 8 < TLEN)
      __builtin_prefetch(xptr + (size_t)(t + 8) * IDIM, 0, 1);

    // ===== layer 1 gates: h0'@Wih1^T + h1@Whh1^T =====
    #pragma unroll
    for (int gi = 0; gi < 4; ++gi) acc[gi] = (v8f){0.f,0.f,0.f,0.f,0.f,0.f,0.f,0.f};
    #pragma unroll
    for (int kb = 0; kb < 2; ++kb) {
      Frag ah;
      const unsigned short* s = &h0A[pw][kb * KBLK + lane * APAD];
      ah.q[0] = *(const uint4*)s; ah.q[1] = *(const uint4*)(s + 8);
      #pragma unroll
      for (int gi = 0; gi < 4; ++gi)
        acc[gi] = __builtin_amdgcn_wmma_f32_16x16x32_bf16(false, ah.v, false, wih1[kb][gi],
                                                          (short)0, acc[gi], false, false);
    }
    #pragma unroll
    for (int kb = 0; kb < 2; ++kb) {
      Frag ah;
      const unsigned short* s = &h1A[pr][kb * KBLK + lane * APAD];
      ah.q[0] = *(const uint4*)s; ah.q[1] = *(const uint4*)(s + 8);
      #pragma unroll
      for (int gi = 0; gi < 4; ++gi)
        acc[gi] = __builtin_amdgcn_wmma_f32_16x16x32_bf16(false, ah.v, false, whh1[kb][gi],
                                                          (short)0, acc[gi], false, false);
    }

    // stage x[t+1] (xA read for step t finished before the mid-step barrier)
    if (t + 1 < TLEN) {
      unsigned int p = (unsigned int)f2bf(xv.x) | ((unsigned int)f2bf(xv.y) << 16);
      *(unsigned int*)&xA[xIdx] = p;
    }

    // elementwise LSTM cell, layer 1
    const bool last = (t == TLEN - 1);
    #pragma unroll
    for (int v = 0; v < 8; ++v) {
      float ig = sigm_b(acc[0][v], sb1[0]);
      float fg = sigm_b(acc[1][v], sb1[1]);
      float gg = tanh_b(acc[2][v], sb1[2]);
      float og = sigm_b(acc[3][v], sb1[3]);
      float c  = fg * c1v[v] + ig * gg;
      c1v[v]   = c;
      float h  = og * tanh_f(c);
      h1A[pw][stBase + v * APAD] = f2bf(h);
      if (last) hOut[(v + 8 * (lane >> 4)) * 64 + j * 16 + (lane & 15)] = h;
    }
    __syncthreads();   // h0A/h1A[pw] + xA ready for next step (also covers hOut)
  }

  // ===== final FC: out[b] = h1_last[b,:] . Wfc + bfc =====
  if (tid < 16) {
    float s = bfc[0];
    for (int k = 0; k < HDIM; ++k) s += hOut[tid * 64 + k] * Wfc[k];
    out[b0 + tid] = s;
  }
#endif
}

extern "C" void kernel_launch(void* const* d_in, const int* in_sizes, int n_in,
                              void* d_out, int out_size, void* d_ws, size_t ws_size,
                              hipStream_t stream) {
  (void)in_sizes; (void)n_in; (void)out_size; (void)d_ws; (void)ws_size;
  const float* x    = (const float*)d_in[0];
  const float* Wih0 = (const float*)d_in[1];
  const float* Whh0 = (const float*)d_in[2];
  const float* bih0 = (const float*)d_in[3];
  const float* bhh0 = (const float*)d_in[4];
  const float* Wih1 = (const float*)d_in[5];
  const float* Whh1 = (const float*)d_in[6];
  const float* bih1 = (const float*)d_in[7];
  const float* bhh1 = (const float*)d_in[8];
  const float* Wfc  = (const float*)d_in[9];
  const float* bfc  = (const float*)d_in[10];
  hipLaunchKernelGGL(lstm_fused_kernel, dim3(1024 / 16), dim3(128), 0, stream,
                     x, Wih0, Whh0, bih0, bhh0, Wih1, Whh1, bih1, bhh1, Wfc, bfc,
                     (float*)d_out);
}
